// ChunkStickyRouter_57226144252170
// MI455X (gfx1250) — matrix operand order repaired and checked by reading, not compiled, not run on hardware
//
#include <hip/hip_runtime.h>
#include <hip/hip_bf16.h>

#define B_ 8
#define S_ 4096
#define D_ 2048
#define H_ 1024
#define E_ 8
#define CHUNK_ 128
#define C_ (S_ / CHUNK_)   // 32 chunks per batch
#define TAU_ 0.7f

#define NTILE 256          // N columns (of H) per block
#define KT 64              // K staged per LDS tile (fallback path)
#define LDA 72             // A row stride in halfs (fallback path)
#define LDB 72             // Bt row stride in halfs (fallback path)

typedef __attribute__((ext_vector_type(16))) __bf16 v16bf;
typedef __attribute__((ext_vector_type(8)))  __bf16 v8bf;
typedef __attribute__((ext_vector_type(2)))  __bf16 v2bf;
typedef __attribute__((ext_vector_type(8)))  float  v8f;

// f32 -> bf16 bits, round-to-nearest-even (bit-twiddle; always available)
__device__ __forceinline__ unsigned short f2bf_bits(float f) {
  union { float f; unsigned u; } a; a.f = f;
  unsigned r = a.u + 0x7fffu + ((a.u >> 16) & 1u);
  return (unsigned short)(r >> 16);
}
__device__ __forceinline__ __bf16 f2bf(float f) {
  union { unsigned short s; __bf16 b; } c; c.s = f2bf_bits(f);
  return c.b;
}
// pack two f32 -> one dword of 2 bf16 (lo in [15:0])
__device__ __forceinline__ unsigned pk_bf16(float lo, float hi) {
#if __has_builtin(__builtin_amdgcn_cvt_pk_bf16_f32)
  union { v2bf v; unsigned u; } c;
  c.v = __builtin_amdgcn_cvt_pk_bf16_f32(lo, hi);
  return c.u;
#else
  return ((unsigned)f2bf_bits(hi) << 16) | (unsigned)f2bf_bits(lo);
#endif
}

// ---------------------------------------------------------------------------
// Prep 1: W1 [D][H] f32 -> W1T [H][D] bf16 (LDS tile transpose)
// grid = (D/64, H/64), block = 256
// ---------------------------------------------------------------------------
__global__ __launch_bounds__(256)
void w1_transpose_kernel(const float* __restrict__ W1, unsigned short* __restrict__ W1T) {
  __shared__ unsigned short t[64 * 72];
  const int tid = threadIdx.x;
  const int k0 = blockIdx.x * 64;   // along D
  const int n0 = blockIdx.y * 64;   // along H
  #pragma unroll
  for (int j = 0; j < 4; ++j) {
    int idx = tid + j * 256;        // 0..1023 : 64 rows x 16 float4
    int kk  = idx >> 4;
    int nn  = (idx & 15) << 2;
    const float4 f = *(const float4*)(W1 + (k0 + kk) * H_ + n0 + nn);
    unsigned short* p = &t[kk * 72 + nn];
    p[0] = f2bf_bits(f.x); p[1] = f2bf_bits(f.y);
    p[2] = f2bf_bits(f.z); p[3] = f2bf_bits(f.w);
  }
  __syncthreads();
  const int nn  = tid >> 2;
  const int kkb = (tid & 3) << 4;
  unsigned up[8];
  #pragma unroll
  for (int q = 0; q < 8; ++q) {
    unsigned lo = t[(kkb + 2 * q + 0) * 72 + nn];
    unsigned hi = t[(kkb + 2 * q + 1) * 72 + nn];
    up[q] = lo | (hi << 16);
  }
  unsigned* dst = (unsigned*)(W1T + (n0 + nn) * D_ + k0 + kkb);
  *(uint4*)(dst)     = make_uint4(up[0], up[1], up[2], up[3]);
  *(uint4*)(dst + 4) = make_uint4(up[4], up[5], up[6], up[7]);
}

// ---------------------------------------------------------------------------
// Prep 2: x f32 -> bf16 (streaming elementwise; 8 elems / thread)
// ---------------------------------------------------------------------------
__global__ __launch_bounds__(256)
void x_convert_kernel(const float* __restrict__ x, unsigned short* __restrict__ xbf) {
  const long i = ((long)blockIdx.x * 256 + threadIdx.x) * 8;
  const float4 f0 = *(const float4*)(x + i);
  const float4 f1 = *(const float4*)(x + i + 4);
  uint4 u;
  u.x = pk_bf16(f0.x, f0.y);
  u.y = pk_bf16(f0.z, f0.w);
  u.z = pk_bf16(f1.x, f1.y);
  u.w = pk_bf16(f1.z, f1.w);
  *(uint4*)(xbf + i) = u;
}

// ---------------------------------------------------------------------------
// Shared epilogue: relu(acc + b1[h]) dotted with W2[h][:], reduce, atomics.
// C layout: VGPR j, lanes 0-15: (M=j, N=lane); lanes 16-31: (M=8+j, N=lane-16).
// ---------------------------------------------------------------------------
__device__ __forceinline__ void router_epilogue(v8f acc[2][8], int n0, int Nbase,
                                                int l15, int lane, int chunk,
                                                const float* __restrict__ b1,
                                                const float* __restrict__ W2,
                                                float* __restrict__ chunk_sums) {
  float contrib[E_];
  #pragma unroll
  for (int e = 0; e < E_; ++e) contrib[e] = 0.f;

  #pragma unroll
  for (int ni = 0; ni < 8; ++ni) {
    const int h = n0 + Nbase + ni * 16 + l15;   // same h for all 16 row-slots of this lane
    const float bb = b1[h];
    float partial = 0.f;
    #pragma unroll
    for (int mi = 0; mi < 2; ++mi)
      #pragma unroll
      for (int j = 0; j < 8; ++j)
        partial += fmaxf(acc[mi][ni][j] + bb, 0.f);
    #pragma unroll
    for (int e = 0; e < E_; ++e)
      contrib[e] += partial * W2[h * E_ + e];
  }

  #pragma unroll
  for (int e = 0; e < E_; ++e)
    #pragma unroll
    for (int m = 16; m >= 1; m >>= 1)
      contrib[e] += __shfl_xor(contrib[e], m, 32);

  if (lane == 0) {
    #pragma unroll
    for (int e = 0; e < E_; ++e)
      atomicAdd(&chunk_sums[chunk * E_ + e], contrib[e]);
  }
}

// ---------------------------------------------------------------------------
// FAST PATH (templated): barrier-free fused GEMM.
//   XBF=true : A from pre-converted bf16 x (pure b128 fragment loads, ~zero VALU)
//   XBF=false: A from f32 x with inline pack
// B always from pre-converted bf16 W1T[H][D] (direct 32B fragment loads, L2-resident).
// grid = (B*C, H/NTILE), block = 256 (8 waves).
// ---------------------------------------------------------------------------
template <bool XBF>
__global__ __launch_bounds__(256)
void router_gemm_fast(const void* __restrict__ xin,
                      const unsigned short* __restrict__ W1T,
                      const float* __restrict__ b1,
                      const float* __restrict__ W2,
                      float* __restrict__ chunk_sums) {
  const int tid  = threadIdx.x;
  const int lane = tid & 31;
  const int wave = tid >> 5;
  const int lgrp = lane >> 4;
  const int l15  = lane & 15;

  const int chunk = blockIdx.x;
  const int r0 = chunk * CHUNK_;
  const int n0 = blockIdx.y * NTILE;
  const int Mbase = (wave & 3) * 32;
  const int Nbase = (wave >> 2) * 128;

  v8f acc[2][8];
  const v8f vzero = {0.f, 0.f, 0.f, 0.f, 0.f, 0.f, 0.f, 0.f};
  #pragma unroll
  for (int i = 0; i < 2; ++i)
    #pragma unroll
    for (int j = 0; j < 8; ++j) acc[i][j] = vzero;

  // A frag (16x32 bf16): lane row = Mbase + mi*16 + l15; K = k0 + 16*lgrp + 0..15
  const float*  paf = (const float*)xin  + (long)(r0 + Mbase + l15) * D_ + lgrp * 16;
  const __bf16* pab = (const __bf16*)xin + (long)(r0 + Mbase + l15) * D_ + lgrp * 16;
  // B frag (32x16 bf16): lane col = n0 + Nbase + ni*16 + l15; K = k0 + 16*lgrp + 0..15
  const __bf16* pb = (const __bf16*)W1T + (long)(n0 + Nbase + l15) * D_ + lgrp * 16;

  for (int k0 = 0; k0 < D_; k0 += 32) {
    union { v16bf v; v8bf h[2]; unsigned u[8]; } afr[2];
    #pragma unroll
    for (int mi = 0; mi < 2; ++mi) {
      if (XBF) {
        const __bf16* p = pab + mi * (16 * D_) + k0;
        afr[mi].h[0] = *(const v8bf*)(p);        // K offsets 0..7
        afr[mi].h[1] = *(const v8bf*)(p + 8);    // K offsets 8..15
      } else {
        const float* p = paf + mi * (16 * D_) + k0;
        const float4 f0 = *(const float4*)(p);
        const float4 f1 = *(const float4*)(p + 4);
        const float4 f2 = *(const float4*)(p + 8);
        const float4 f3 = *(const float4*)(p + 12);
        afr[mi].u[0] = pk_bf16(f0.x, f0.y);
        afr[mi].u[1] = pk_bf16(f0.z, f0.w);
        afr[mi].u[2] = pk_bf16(f1.x, f1.y);
        afr[mi].u[3] = pk_bf16(f1.z, f1.w);
        afr[mi].u[4] = pk_bf16(f2.x, f2.y);
        afr[mi].u[5] = pk_bf16(f2.z, f2.w);
        afr[mi].u[6] = pk_bf16(f3.x, f3.y);
        afr[mi].u[7] = pk_bf16(f3.z, f3.w);
      }
    }
    #pragma unroll
    for (int ni = 0; ni < 8; ++ni) {
      const v16bf bfr = *(const v16bf*)(pb + ni * (16 * D_) + k0);  // 32B, 2x b128
      acc[0][ni] = __builtin_amdgcn_wmma_f32_16x16x32_bf16(
          false, afr[0].v, false, bfr, (short)0, acc[0][ni], false, false);
      acc[1][ni] = __builtin_amdgcn_wmma_f32_16x16x32_bf16(
          false, afr[1].v, false, bfr, (short)0, acc[1][ni], false, false);
    }
  }

  router_epilogue(acc, n0, Nbase, l15, lane, chunk, b1, W2, chunk_sums);
}

// ---------------------------------------------------------------------------
// FALLBACK PATH (tiny workspace): LDS-staged version.
// ---------------------------------------------------------------------------
__global__ __launch_bounds__(256)
void router_gemm_lds(const float* __restrict__ x,
                     const float* __restrict__ W1,
                     const float* __restrict__ b1,
                     const float* __restrict__ W2,
                     float* __restrict__ chunk_sums) {
  __shared__ __bf16 lds_a[CHUNK_ * LDA];
  __shared__ __bf16 lds_bT[NTILE * LDB];

  const int tid  = threadIdx.x;
  const int lane = tid & 31;
  const int wave = tid >> 5;
  const int lgrp = lane >> 4;
  const int l15  = lane & 15;

  const int chunk = blockIdx.x;
  const int r0 = chunk * CHUNK_;
  const int n0 = blockIdx.y * NTILE;
  const int Mbase = (wave & 3) * 32;
  const int Nbase = (wave >> 2) * 128;

  v8f acc[2][8];
  const v8f vzero = {0.f, 0.f, 0.f, 0.f, 0.f, 0.f, 0.f, 0.f};
  #pragma unroll
  for (int i = 0; i < 2; ++i)
    #pragma unroll
    for (int j = 0; j < 8; ++j) acc[i][j] = vzero;

  for (int k0 = 0; k0 < D_; k0 += KT) {
    #pragma unroll
    for (int j = 0; j < 8; ++j) {
      int idx = tid + j * 256;
      int row = idx >> 4;
      int col = (idx & 15) << 2;
      const float4 f = *(const float4*)(x + (r0 + row) * D_ + k0 + col);
      __bf16* p = &lds_a[row * LDA + col];
      p[0] = f2bf(f.x); p[1] = f2bf(f.y); p[2] = f2bf(f.z); p[3] = f2bf(f.w);
    }
    #pragma unroll
    for (int j = 0; j < 16; ++j) {
      int idx = tid + j * 256;
      int kk  = idx >> 6;
      int nn  = (idx & 63) << 2;
      const float4 f = *(const float4*)(W1 + (k0 + kk) * H_ + n0 + nn);
      lds_bT[(nn + 0) * LDB + kk] = f2bf(f.x);
      lds_bT[(nn + 1) * LDB + kk] = f2bf(f.y);
      lds_bT[(nn + 2) * LDB + kk] = f2bf(f.z);
      lds_bT[(nn + 3) * LDB + kk] = f2bf(f.w);
    }
    __syncthreads();

    #pragma unroll
    for (int ks = 0; ks < KT; ks += 32) {
      union { v16bf v; v8bf h[2]; } afr[2];
      #pragma unroll
      for (int mi = 0; mi < 2; ++mi) {
        const __bf16* p = &lds_a[(Mbase + mi * 16 + l15) * LDA + ks + lgrp * 8];
        afr[mi].h[0] = *(const v8bf*)(p);
        afr[mi].h[1] = *(const v8bf*)(p + 16);
      }
      #pragma unroll
      for (int ni = 0; ni < 8; ++ni) {
        union { v16bf v; v8bf h[2]; } bfr;
        const __bf16* p = &lds_bT[(Nbase + ni * 16 + l15) * LDB + ks + lgrp * 8];
        bfr.h[0] = *(const v8bf*)(p);
        bfr.h[1] = *(const v8bf*)(p + 16);
        acc[0][ni] = __builtin_amdgcn_wmma_f32_16x16x32_bf16(
            false, afr[0].v, false, bfr.v, (short)0, acc[0][ni], false, false);
        acc[1][ni] = __builtin_amdgcn_wmma_f32_16x16x32_bf16(
            false, afr[1].v, false, bfr.v, (short)0, acc[1][ni], false, false);
      }
    }
    __syncthreads();
  }

  router_epilogue(acc, n0, Nbase, l15, lane, chunk, b1, W2, chunk_sums);
}

// ---------------------------------------------------------------------------
__global__ void zero_sums_kernel(float* s) {
  int i = blockIdx.x * 256 + threadIdx.x;
  if (i < B_ * C_ * E_) s[i] = 0.f;
}

// ---------------------------------------------------------------------------
__global__ __launch_bounds__(256)
void router_finalize_kernel(const float* __restrict__ sums,
                            const float* __restrict__ b2,
                            float* __restrict__ out,
                            int* __restrict__ idx_ws) {
  __shared__ float cl[B_ * C_ * E_];
  const int tid = threadIdx.x;
  for (int i = tid; i < B_ * C_ * E_; i += 256)
    cl[i] = sums[i] * (1.0f / (float)CHUNK_) + b2[i & (E_ - 1)];
  __syncthreads();

  if (tid < B_) {
    const int b = tid;
    const float* p0 = &cl[(b * C_) * E_];
    int prev = 0;
    float best = p0[0];
    for (int e = 1; e < E_; ++e)
      if (p0[e] > best) { best = p0[e]; prev = e; }   // first-max (jnp.argmax)
    idx_ws[b * C_] = prev;
    out[(long)B_ * S_ * E_ + b * C_] = (float)prev;

    for (int c = 1; c < C_; ++c) {
      const float* p = &cl[(b * C_ + c) * E_];
      int ce = 0;
      float cb = p[0];
      for (int e = 1; e < E_; ++e)
        if (p[e] > cb) { cb = p[e]; ce = e; }
      if (cb - p[prev] > TAU_) prev = ce;             // hysteresis, strict >
      idx_ws[b * C_ + c] = prev;
      out[(long)B_ * S_ * E_ + b * C_ + c] = (float)prev;
    }
  }
}

// ---------------------------------------------------------------------------
__global__ __launch_bounds__(256)
void router_weights_kernel(const int* __restrict__ idx_ws,
                           float* __restrict__ out) {
  const int g = blockIdx.x * 256 + threadIdx.x;  // flat row 0..32767
  const int e = idx_ws[g >> 7];
  float4 lo, hi;
  lo.x = (e == 0) ? 1.f : 0.f;  lo.y = (e == 1) ? 1.f : 0.f;
  lo.z = (e == 2) ? 1.f : 0.f;  lo.w = (e == 3) ? 1.f : 0.f;
  hi.x = (e == 4) ? 1.f : 0.f;  hi.y = (e == 5) ? 1.f : 0.f;
  hi.z = (e == 6) ? 1.f : 0.f;  hi.w = (e == 7) ? 1.f : 0.f;
  float* p = out + (long)g * E_;
  *(float4*)(p) = lo;
  *(float4*)(p + 4) = hi;
}

// ---------------------------------------------------------------------------
extern "C" void kernel_launch(void* const* d_in, const int* in_sizes, int n_in,
                              void* d_out, int out_size, void* d_ws, size_t ws_size,
                              hipStream_t stream) {
  const float* x  = (const float*)d_in[0];   // [8,4096,2048]
  const float* W1 = (const float*)d_in[1];   // [2048,1024]
  const float* b1 = (const float*)d_in[2];   // [1024]
  const float* W2 = (const float*)d_in[3];   // [1024,8]
  const float* b2 = (const float*)d_in[4];   // [8]
  float* out = (float*)d_out;

  const size_t W1T_BYTES = (size_t)H_ * D_ * sizeof(unsigned short);       // 4 MiB
  const size_t XBF_BYTES = (size_t)B_ * S_ * D_ * sizeof(unsigned short);  // 128 MiB

  float* ws_sums = (float*)d_ws;                                     // 2048 f32
  int*   ws_idx  = (int*)((char*)d_ws + B_ * C_ * E_ * sizeof(float));
  unsigned short* ws_w1t = (unsigned short*)((char*)d_ws + 16384);
  unsigned short* ws_xbf = (unsigned short*)((char*)d_ws + 16384 + W1T_BYTES);

  zero_sums_kernel<<<(B_ * C_ * E_ + 255) / 256, 256, 0, stream>>>(ws_sums);

  dim3 grid(B_ * C_, H_ / NTILE);            // 256 x 4
  dim3 tgrid(D_ / 64, H_ / 64);              // 32 x 16

  if (ws_size >= 16384 + W1T_BYTES + XBF_BYTES) {
    // Tier 1: pre-convert both operands; GEMM loop is pure b128 loads + WMMA.
    w1_transpose_kernel<<<tgrid, 256, 0, stream>>>(W1, ws_w1t);
    x_convert_kernel<<<(int)((size_t)B_ * S_ * D_ / (8 * 256)), 256, 0, stream>>>(x, ws_xbf);
    router_gemm_fast<true><<<grid, 256, 0, stream>>>(ws_xbf, ws_w1t, b1, W2, ws_sums);
  } else if (ws_size >= 16384 + W1T_BYTES) {
    // Tier 2: pre-convert W1 only; A packed inline from f32.
    w1_transpose_kernel<<<tgrid, 256, 0, stream>>>(W1, ws_w1t);
    router_gemm_fast<false><<<grid, 256, 0, stream>>>(x, ws_w1t, b1, W2, ws_sums);
  } else {
    // Tier 3: LDS-staged fallback.
    router_gemm_lds<<<grid, 256, 0, stream>>>(x, W1, b1, W2, ws_sums);
  }

  router_finalize_kernel<<<1, 256, 0, stream>>>(ws_sums, b2, out, ws_idx);
  router_weights_kernel<<<(B_ * S_) / 256, 256, 0, stream>>>(ws_idx, out);
}
